// NCENewAverage_1846835937452
// MI455X (gfx1250) — compile-verified, matrix operand order checked by reading, not run on hardware
//
#include <hip/hip_runtime.h>
#include <hip/hip_bf16.h>

typedef __attribute__((ext_vector_type(16))) _Float16 v16h;
typedef __attribute__((ext_vector_type(8)))  float    v8f;

// Problem constants (match reference: B=128, D=128, NEG=4096, K=512, T=0.07)
#define NB     128
#define ND     128
#define NEGP1  4097          // NEG + 1
#define NSORT  4096
#define KH     256           // K/2
#define TAIL0  3584          // NEG - K
#define OUTW   513           // K + 1
#define INVT   14.2857142857142857f

__global__ __launch_bounds__(256) void nce_main_kernel(
    const float* __restrict__ x, const int* __restrict__ y,
    const int* __restrict__ idx, const float* __restrict__ memory,
    float* __restrict__ out, float* __restrict__ rowsum)
{
    __shared__ float     scores[NEGP1 + 3];   // [0]=positive, [1..4096]=negatives
    __shared__ _Float16  x_sh[ND];
    __shared__ float     red[256];

    const int b    = blockIdx.x;
    const int tid  = threadIdx.x;
    const int wid  = tid >> 5;
    const int lane = tid & 31;
    const int m    = lane & 15;       // A-matrix row within tile
    const int half = lane >> 4;       // 0: K offsets {0..7,16..23}; 1: {8..15,24..31}
    const int kbase = half * 8;

    // stage x[b] into LDS as f16 (B operand source)
    if (tid < ND) x_sh[tid] = (_Float16)x[b * ND + tid];
    __syncthreads();

    // ---- Phase 1: WMMA scoring. Each wave handles 16 gathered rows per tile. ----
    for (int t = wid; t < 257; t += 8) {          // 257*16 = 4112 covers 4097 rows
        const int k0 = t * 16;
        const int k  = k0 + m;
        int src;
        if (k == 0)            src = y[b];              // positive slot
        else if (k < NEGP1)    src = idx[b * NEGP1 + k];
        else                   src = 0;                 // padded lane, discarded
        const float* p = memory + (size_t)src * ND + kbase;

        // prefetch the row this lane will gather 8 tiles from now
        {
            const int k2 = k + 128;
            const int src2 = (k2 < NEGP1) ? idx[b * NEGP1 + k2] : 0;
            __builtin_prefetch(memory + (size_t)src2 * ND, 0, 1);
        }

        v8f c = {0.f, 0.f, 0.f, 0.f, 0.f, 0.f, 0.f, 0.f};
        #pragma unroll
        for (int s = 0; s < 4; ++s) {                 // K = 128 in 4 chunks of 32
            const float4* q0 = (const float4*)(p + s * 32);      // K = s*32+kbase+0..7
            const float4* q1 = (const float4*)(p + s * 32 + 16); // K = s*32+kbase+16..23
            float4 f0 = q0[0], f1 = q0[1];
            float4 f2 = q1[0], f3 = q1[1];
            v16h a;
            a[0]  = (_Float16)f0.x; a[1]  = (_Float16)f0.y;
            a[2]  = (_Float16)f0.z; a[3]  = (_Float16)f0.w;
            a[4]  = (_Float16)f1.x; a[5]  = (_Float16)f1.y;
            a[6]  = (_Float16)f1.z; a[7]  = (_Float16)f1.w;
            a[8]  = (_Float16)f2.x; a[9]  = (_Float16)f2.y;
            a[10] = (_Float16)f2.z; a[11] = (_Float16)f2.w;
            a[12] = (_Float16)f3.x; a[13] = (_Float16)f3.y;
            a[14] = (_Float16)f3.z; a[15] = (_Float16)f3.w;

            // B = x replicated across all 16 N-columns: same per-lane K map as A
            const _Float16* xk = x_sh + s * 32 + kbase;
            v16h bb;
            #pragma unroll
            for (int j = 0; j < 8; ++j) { bb[j] = xk[j]; bb[8 + j] = xk[16 + j]; }

            c = __builtin_amdgcn_wmma_f32_16x16x32_f16(
                    /*neg_a=*/false, a, /*neg_b=*/false, bb,
                    /*c_mod=*/(short)0, c, /*reuse_a=*/false, /*reuse_b=*/false);
        }

        // All C columns are identical; lane 0 holds rows 0..7, lane 16 rows 8..15.
        if (m == 0) {
            const int kb = k0 + half * 8;
            #pragma unroll
            for (int v = 0; v < 8; ++v) {
                const int kk = kb + v;
                if (kk < NEGP1) scores[kk] = __expf(c[v] * INVT);
            }
        }
    }
    __syncthreads();

    // ---- Phase 2: bitonic sort scores[1..4096] descending (values only) ----
    float* s = scores + 1;
    for (int size = 2; size <= NSORT; size <<= 1) {
        for (int stride = size >> 1; stride > 0; stride >>= 1) {
            __syncthreads();
            for (int t2 = tid; t2 < (NSORT / 2); t2 += 256) {
                const int lo = ((t2 & ~(stride - 1)) << 1) | (t2 & (stride - 1));
                const int hi = lo + stride;
                const bool desc = ((lo & size) == 0);
                const float a0 = s[lo], a1 = s[hi];
                const bool doswap = desc ? (a0 < a1) : (a0 > a1);
                if (doswap) { s[lo] = a1; s[hi] = a0; }
            }
        }
    }
    __syncthreads();

    // ---- Phase 3: emit selected values (unscaled) + deterministic row sum ----
    const float v1 = s[tid];           // hardest ranks 0..255
    const float v2 = s[TAIL0 + tid];   // tail-band ranks 3584..3839
    out[b * OUTW + 1 + tid]        = v1;
    out[b * OUTW + 1 + KH + tid]   = v2;
    float r = v1 + v2;
    if (tid == 0) { out[b * OUTW] = scores[0]; r += scores[0]; }
    red[tid] = r;
    __syncthreads();
    #pragma unroll
    for (int off = 128; off > 0; off >>= 1) {
        if (tid < off) red[tid] += red[tid + off];
        __syncthreads();
    }
    if (tid == 0) rowsum[b] = red[0];
}

__global__ __launch_bounds__(128) void nce_z_kernel(
    const float* __restrict__ rowsum, float* __restrict__ invZ, float nrows)
{
    __shared__ float red[NB];
    const int t = threadIdx.x;
    red[t] = rowsum[t];
    __syncthreads();
    #pragma unroll
    for (int off = NB / 2; off > 0; off >>= 1) {
        if (t < off) red[t] += red[t + off];
        __syncthreads();
    }
    if (t == 0) {
        const float Z = (red[0] / (float)(NB * OUTW)) * nrows;
        *invZ = 1.0f / Z;
    }
}

__global__ __launch_bounds__(256) void nce_scale_kernel(
    float* __restrict__ out, const float* __restrict__ invZ, int n)
{
    const int i = blockIdx.x * blockDim.x + threadIdx.x;
    if (i < n) out[i] *= *invZ;
}

extern "C" void kernel_launch(void* const* d_in, const int* in_sizes, int n_in,
                              void* d_out, int out_size, void* d_ws, size_t ws_size,
                              hipStream_t stream) {
    const float* x      = (const float*)d_in[0];
    const int*   y      = (const int*)  d_in[1];
    const int*   idx    = (const int*)  d_in[2];
    const float* memory = (const float*)d_in[3];
    float* out = (float*)d_out;

    float* rowsum = (float*)d_ws;          // 128 floats
    float* invZ   = rowsum + NB;           // 1 float

    const int nrows = in_sizes[3] / ND;    // memory bank size N

    nce_main_kernel<<<NB, 256, 0, stream>>>(x, y, idx, memory, out, rowsum);
    nce_z_kernel<<<1, NB, 0, stream>>>(rowsum, invZ, (float)nrows);
    nce_scale_kernel<<<(out_size + 255) / 256, 256, 0, stream>>>(out, invZ, out_size);
}
